// SimpleGRU_83279415869868
// MI455X (gfx1250) — matrix operand order repaired and checked by reading, not compiled, not run on hardware
//
#include <hip/hip_runtime.h>
#include <hip/hip_bf16.h>

typedef __attribute__((ext_vector_type(16))) _Float16 v16h;
typedef __attribute__((ext_vector_type(8)))  float    v8f;

#define GRU_T 2048
#define GRU_H 32
#define GRU_B 4096

__device__ __forceinline__ float fast_tanh(float t) {
#if __has_builtin(__builtin_amdgcn_tanhf)
    return __builtin_amdgcn_tanhf(t);          // v_tanh_f32 (single TRANS op)
#else
    // tanh(t) = 1 - 2/(e^{2t}+1)
    return 1.0f - 2.0f * __builtin_amdgcn_rcpf(1.0f + __expf(2.0f * t));
#endif
}

// sigmoid(y) = 0.5*tanh(y/2) + 0.5 ; the /2 is pre-folded into weights/biases,
// so the caller passes a = y/2 and this is just tanh + one fma.
__device__ __forceinline__ float sigm_from_half(float a) {
    return fmaf(0.5f, fast_tanh(a), 0.5f);
}

// One wave (32 lanes) owns 16 batch rows for the whole T=2048 recurrence.
// Per step: gh = W_hh_tile(16x32,f16) x h(32x16,f16) + bias  via 6x
// v_wmma_f32_16x16x32_f16; gates are per-lane f32 VALU (native v_tanh_f32);
// new h packs directly into the WMMA B-operand layout (no cross-lane moves).
__global__ __launch_bounds__(32) void SimpleGRU_wmma_gfx1250(
    const float* __restrict__ x,     // (B, T)  (I == 1)
    const float* __restrict__ W_ih,  // (96, 1)
    const float* __restrict__ W_hh,  // (96, 32)
    const float* __restrict__ b_ih,  // (96)
    const float* __restrict__ b_hh,  // (96)
    const float* __restrict__ W_fc,  // (2, 32)
    const float* __restrict__ b_fc,  // (2)
    float* __restrict__ out)         // (B, 2)
{
    const int lane  = threadIdx.x & 31;
    const int m     = lane & 15;   // batch column within tile / A-matrix row
    const int hi    = lane >> 4;   // which K/position half this lane carries
    const int batch = blockIdx.x * 16 + m;

    // ---- Constant A tiles (W_hh, f16) and bias-folded C-init tiles ----
    // Gate g in {r,z,n}, sub-tile tt in {0,1}; row permutation inside a gate:
    // rowpos(tt, M) = M + 8*tt + (M>=8 ? 8 : 0)  — chosen so the D-tile layout
    // of h_new coincides with the B-operand layout for the next step.
    // r/z tiles and biases are pre-scaled by 0.5 (sigmoid-as-tanh folding).
    v16h A[6];
    v8f  Cinit[6];
#pragma unroll
    for (int g = 0; g < 3; ++g) {
        const float gs = (g < 2) ? 0.5f : 1.0f;   // half-scale for sigmoid gates
#pragma unroll
        for (int tt = 0; tt < 2; ++tt) {
            const int tile   = g * 2 + tt;
            const int rowsel = m + 8 * tt + ((m >= 8) ? 8 : 0);
#pragma unroll
            for (int e = 0; e < 16; ++e) {
                const int K = hi * 16 + e; // A 16x32 f16 layout
                A[tile][e] = (_Float16)(gs * W_hh[(g * GRU_H + rowsel) * GRU_H + K]);
            }
#pragma unroll
            for (int r = 0; r < 8; ++r) {
                const int M  = hi ? (8 + r) : r;                  // D-tile row per lane-half
                const int rp = M + 8 * tt + ((M >= 8) ? 8 : 0);
                const int gi = g * GRU_H + rp;
                float bb = b_hh[gi];
                if (g < 2) bb = 0.5f * (bb + b_ih[gi]);           // fold both biases, half-scaled
                Cinit[tile][r] = bb;                              // n keeps b_ih_n outside r*hn
            }
        }
    }

    // ---- Per-lane input-gate constants (I==1 so gx = x*W_ih + b_ih) ----
    float Wr[16], Wz[16], Wn[16], Bn[16];
#pragma unroll
    for (int p = 0; p < 16; ++p) {
        const int pos = hi * 16 + p;
        Wr[p] = 0.5f * W_ih[0 * GRU_H + pos];   // half-scaled for sigmoid gates
        Wz[p] = 0.5f * W_ih[1 * GRU_H + pos];
        Wn[p] = W_ih[2 * GRU_H + pos];
        Bn[p] = b_ih[2 * GRU_H + pos];
    }

    // h state lives directly in WMMA B-operand layout: lane element p holds
    // h[batch, hi*16 + p] as f16.
    v16h hreg;
#pragma unroll
    for (int p = 0; p < 16; ++p) hreg[p] = (_Float16)0.0f;

    const float* xrow = x + (size_t)batch * GRU_T;

    for (int t0 = 0; t0 < GRU_T; t0 += 4) {
        const float4 xv = *(const float4*)(xrow + t0);
        __builtin_prefetch(xrow + t0 + 64, 0, 1);   // global_prefetch_b8, stream x ahead
#pragma unroll
        for (int s = 0; s < 4; ++s) {
            const float xval = (s == 0) ? xv.x : (s == 1) ? xv.y : (s == 2) ? xv.z : xv.w;

            v8f D[6];
#pragma unroll
            for (int tile = 0; tile < 6; ++tile) {
                D[tile] = __builtin_amdgcn_wmma_f32_16x16x32_f16(
                    /*neg_a=*/false, A[tile], /*neg_b=*/false, hreg,
                    /*c_mod=*/(short)0, Cinit[tile],
                    /*reuse_a=*/false, /*reuse_b=*/false);
            }

            v16h hnew;
#pragma unroll
            for (int p = 0; p < 16; ++p) {
                const float hr = (p < 8) ? D[0][p] : D[1][p - 8];  // = (xr'+hr')/2 pre-act
                const float hz = (p < 8) ? D[2][p] : D[3][p - 8];
                const float hn = (p < 8) ? D[4][p] : D[5][p - 8];  // h@Wn + b_hh_n (unscaled)
                const float r  = sigm_from_half(fmaf(xval, Wr[p], hr));
                const float z  = sigm_from_half(fmaf(xval, Wz[p], hz));
                const float nn = fast_tanh(fmaf(r, hn, fmaf(xval, Wn[p], Bn[p])));
                const float hp = (float)hreg[p];
                hnew[p] = (_Float16)fmaf(z, hp - nn, nn);          // (1-z)*n + z*h
            }
            hreg = hnew;
        }
    }

    // ---- FC head: out[b, c] = sum_u h[b,u] * W_fc[c,u] + b_fc[c], C = 2 ----
    float acc0 = 0.0f, acc1 = 0.0f;
#pragma unroll
    for (int p = 0; p < 16; ++p) {
        const int   pos = hi * 16 + p;
        const float hv  = (float)hreg[p];
        acc0 = fmaf(hv, W_fc[0 * GRU_H + pos], acc0);
        acc1 = fmaf(hv, W_fc[1 * GRU_H + pos], acc1);
    }
    acc0 += __shfl_xor(acc0, 16, 32);   // combine the two position halves
    acc1 += __shfl_xor(acc1, 16, 32);
    if (hi == 0) {
        out[batch * 2 + 0] = acc0 + b_fc[0];
        out[batch * 2 + 1] = acc1 + b_fc[1];
    }
}

extern "C" void kernel_launch(void* const* d_in, const int* in_sizes, int n_in,
                              void* d_out, int out_size, void* d_ws, size_t ws_size,
                              hipStream_t stream) {
    const float* x    = (const float*)d_in[0];
    const float* W_ih = (const float*)d_in[1];
    const float* W_hh = (const float*)d_in[2];
    const float* b_ih = (const float*)d_in[3];
    const float* b_hh = (const float*)d_in[4];
    const float* W_fc = (const float*)d_in[5];
    const float* b_fc = (const float*)d_in[6];
    float* out = (float*)d_out;

    const int nblocks = GRU_B / 16;  // 256 waves, one 32-thread wave per block
    SimpleGRU_wmma_gfx1250<<<nblocks, 32, 0, stream>>>(
        x, W_ih, W_hh, b_ih, b_hh, W_fc, b_fc, out);
}
